// HierarchicalVim_59794534694954
// MI455X (gfx1250) — matrix operand order repaired and checked by reading, not compiled, not run on hardware
//
#include <hip/hip_runtime.h>
#include <hip/hip_bf16.h>
#include <math.h>

typedef __attribute__((ext_vector_type(16))) _Float16 v16h;
typedef __attribute__((ext_vector_type(8)))  float    v8f;

// ---------------------------------------------------------------- helpers
__device__ __forceinline__ float sigmoidf_(float x) { return 1.0f / (1.0f + __expf(-x)); }
__device__ __forceinline__ float siluf_(float x)    { return x * sigmoidf_(x); }
__device__ __forceinline__ float softplusf_(float x){ return (x > 20.f) ? x : log1pf(__expf(x)); }

struct alignas(16) f4v { float x, y, z, w; };
struct alignas(16) h8v { _Float16 h[8]; };

// ---------------------------------------------------------------- WMMA GEMM
// out[M,N] (row stride ldc) (+)= act( A[M,K] (row stride lda) @ W[N,K]^T + bias[N] )
// 256 threads = 8 waves; block tile 64x64; wave (wy,wx) = 16 rows x 32 cols
// = two 16x16 WMMA tiles sharing one A fragment. K step 32.
// Double-buffered LDS: next tile's global loads issue before current tile's
// ds_load+wmma (wmma waits only dscnt), commit (cvt+ds_store) after the wmma.
// All edge handling is branchless via clamped addresses + selects.
template<int ACT, bool ACC>
__global__ __launch_bounds__(256)
void wmma_gemm_kernel(float* __restrict__ out, int ldc,
                      const float* __restrict__ A, int lda,
                      const float* __restrict__ W,
                      const float* __restrict__ bias,
                      int M, int N, int K)
{
    __shared__ __align__(16) _Float16 As[2][64][40];
    __shared__ __align__(16) _Float16 Bs[2][64][40];

    const int tid  = threadIdx.x;
    const int lane = tid & 31;
    const int wave = tid >> 5;      // 0..7
    const int wy   = wave & 3;      // M sub-tile (16 rows)
    const int wx   = wave >> 2;     // N pair (32 cols)
    const int m0   = blockIdx.y * 64;
    const int n0   = blockIdx.x * 64;

    const int mfrag = lane & 15;
    const int g     = lane >> 4;

    // loader mapping: each thread owns one 8-wide row segment per matrix tile
    const int lr  = tid >> 2;        // 0..63 tile row
    const int lcs = (tid & 3) * 8;   // segment start col

    const int gmA = m0 + lr;
    const int gnB = n0 + lr;
    const bool mok = (gmA < M);
    const bool nok = (gnB < N);
    const float* __restrict__ arow = A + (size_t)(mok ? gmA : (M - 1)) * lda;
    const float* __restrict__ brow = W + (size_t)(nok ? gnB : (N - 1)) * K;

    const bool vecok = ((lda & 3) == 0) && ((K & 3) == 0);
    const int  kmain = vecok ? (K & ~31) : 0;
    const int  nk    = (K + 31) >> 5;

    float ra[8], rb[8];

    auto fetch = [&](int k0) {
        if (k0 + 32 <= kmain) {                 // fully vectorizable tile
            const float* sa = arow + k0 + lcs;
            f4v f0 = *(const f4v*)sa, f1 = *(const f4v*)(sa + 4);
            ra[0] = f0.x; ra[1] = f0.y; ra[2] = f0.z; ra[3] = f0.w;
            ra[4] = f1.x; ra[5] = f1.y; ra[6] = f1.z; ra[7] = f1.w;
            const float* sb = brow + k0 + lcs;
            f4v h0 = *(const f4v*)sb, h1 = *(const f4v*)(sb + 4);
            rb[0] = h0.x; rb[1] = h0.y; rb[2] = h0.z; rb[3] = h0.w;
            rb[4] = h1.x; rb[5] = h1.y; rb[6] = h1.z; rb[7] = h1.w;
        } else {                                // branchless scalar tail
#pragma unroll
            for (int j = 0; j < 8; ++j) {
                int gk  = k0 + lcs + j;
                int gkc = (gk < K) ? gk : (K - 1);   // clamped: load always legal
                float av = arow[gkc];
                float bv = brow[gkc];
                ra[j] = (gk < K) ? av : 0.f;
                rb[j] = (gk < K) ? bv : 0.f;
            }
        }
    };
    auto commit = [&](int buf) {
        h8v ta, tb;
#pragma unroll
        for (int j = 0; j < 8; ++j) {
            ta.h[j] = (_Float16)(mok ? ra[j] : 0.f);
            tb.h[j] = (_Float16)(nok ? rb[j] : 0.f);
        }
        *(h8v*)&As[buf][lr][lcs] = ta;
        *(h8v*)&Bs[buf][lr][lcs] = tb;
    };

    v8f c0 = {0.f,0.f,0.f,0.f,0.f,0.f,0.f,0.f};
    v8f c1 = {0.f,0.f,0.f,0.f,0.f,0.f,0.f,0.f};

    fetch(0);
    commit(0);
    int buf = 0;
    for (int t = 0; t < nk; ++t) {
        __syncthreads();
        const bool have = (t + 1 < nk);
        if (have) fetch((t + 1) * 32);          // global loads in flight

        // pack fragments: lane half g holds k = 16g+i for both A and B (contiguous k order)
        v16h a = {}, b0 = {}, b1 = {};
#pragma unroll
        for (int i = 0; i < 16; ++i) {
            int ka = g * 16 + i;
            a[i]  = As[buf][wy * 16 + mfrag][ka];
            int kb = g * 16 + i;
            b0[i] = Bs[buf][wx * 32 + mfrag][kb];
            b1[i] = Bs[buf][wx * 32 + 16 + mfrag][kb];
        }
        c0 = __builtin_amdgcn_wmma_f32_16x16x32_f16(false, a, false, b0, (short)0, c0, false, false);
        c1 = __builtin_amdgcn_wmma_f32_16x16x32_f16(false, a, false, b1, (short)0, c1, false, false);

        if (have) commit(buf ^ 1);              // cvt + ds_store after the wmma
        buf ^= 1;
    }

    // D layout: lane n = mfrag (+16g half), vgpr r -> m = r + 8g
    const int gn0 = n0 + wx * 32 + mfrag;
    const int gn1 = gn0 + 16;
#pragma unroll
    for (int r = 0; r < 8; ++r) {
        int gm = m0 + wy * 16 + r + 8 * g;
        if (gm >= M) continue;
        if (gn0 < N) {
            float v = c0[r];
            if (bias) v += bias[gn0];
            if (ACT == 1) v = softplusf_(v);
            size_t o = (size_t)gm * ldc + gn0;
            if (ACC) out[o] += v; else out[o] = v;
        }
        if (gn1 < N) {
            float v = c1[r];
            if (bias) v += bias[gn1];
            if (ACT == 1) v = softplusf_(v);
            size_t o = (size_t)gm * ldc + gn1;
            if (ACC) out[o] += v; else out[o] = v;
        }
    }
}

static inline void gemm(hipStream_t s, float* out, int ldc, const float* A, int lda,
                        const float* W, const float* bias,
                        int M, int N, int K, int act, bool acc)
{
    dim3 g((N + 63) / 64, (M + 63) / 64), b(256);
    if (acc)            wmma_gemm_kernel<0, true ><<<g, b, 0, s>>>(out, ldc, A, lda, W, bias, M, N, K);
    else if (act == 1)  wmma_gemm_kernel<1, false><<<g, b, 0, s>>>(out, ldc, A, lda, W, bias, M, N, K);
    else                wmma_gemm_kernel<0, false><<<g, b, 0, s>>>(out, ldc, A, lda, W, bias, M, N, K);
}

// ---------------------------------------------------------------- patch embed
__global__ void patch_embed_kernel(float* __restrict__ out,
                                   const float* __restrict__ x,
                                   const float* __restrict__ w,
                                   const float* __restrict__ bias, int B)
{
    int idx = blockIdx.x * blockDim.x + threadIdx.x;
    int total = B * 3136 * 96;
    if (idx >= total) return;
    int d = idx % 96;
    int p = (idx / 96) % 3136;
    int b = idx / (96 * 3136);
    int ph = p / 56, pw = p % 56;
    float acc = bias[d];
    for (int cch = 0; cch < 3; ++cch)
        for (int py = 0; py < 4; ++py)
            for (int qx = 0; qx < 4; ++qx) {
                float xv = x[(((size_t)b * 3 + cch) * 224 + (ph * 4 + py)) * 224 + (pw * 4 + qx)];
                float wv = w[(((size_t)d * 3 + cch) * 4 + py) * 4 + qx];
                acc += xv * wv;
            }
    out[idx] = acc;
}

// ---------------------------------------------------------------- norms (wave32 per row)
__global__ void layernorm_kernel(float* __restrict__ x, const float* __restrict__ w,
                                 const float* __restrict__ b, int rows, int C)
{
    int row = blockIdx.x;
    if (row >= rows) return;
    int lane = threadIdx.x;
    float* xr = x + (size_t)row * C;
    float s = 0.f;
    for (int c = lane; c < C; c += 32) s += xr[c];
    for (int o = 16; o > 0; o >>= 1) s += __shfl_xor(s, o, 32);
    float mean = s / C;
    float v = 0.f;
    for (int c = lane; c < C; c += 32) { float d = xr[c] - mean; v += d * d; }
    for (int o = 16; o > 0; o >>= 1) v += __shfl_xor(v, o, 32);
    float rstd = rsqrtf(v / C + 1e-5f);
    for (int c = lane; c < C; c += 32) xr[c] = (xr[c] - mean) * rstd * w[c] + b[c];
}

__global__ void rmsnorm_kernel(float* __restrict__ out, const float* __restrict__ x,
                               const float* __restrict__ w, int rows, int C)
{
    int row = blockIdx.x;
    if (row >= rows) return;
    int lane = threadIdx.x;
    const float* xr = x + (size_t)row * C;
    float s = 0.f;
    for (int c = lane; c < C; c += 32) { float v = xr[c]; s += v * v; }
    for (int o = 16; o > 0; o >>= 1) s += __shfl_xor(s, o, 32);
    float r = rsqrtf(s / C + 1e-5f);
    float* orow = out + (size_t)row * C;
    for (int c = lane; c < C; c += 32) orow[c] = xr[c] * r * w[c];
}

// ---------------------------------------------------------------- misc elementwise
__global__ void residual_kernel(float* __restrict__ res, const float* __restrict__ cur,
                                int n, int first)
{
    int i = blockIdx.x * blockDim.x + threadIdx.x;
    if (i < n) res[i] = first ? cur[i] : (res[i] + cur[i]);
}

// depthwise causal conv (kernel 4) + SiLU; rev==1 processes the flipped sequence
__global__ void conv_silu_kernel(float* __restrict__ xc, const float* __restrict__ xz,
                                 int ldxz, const float* __restrict__ w,
                                 const float* __restrict__ bias,
                                 int B, int L, int di, int rev)
{
    int idx = blockIdx.x * blockDim.x + threadIdx.x;
    int total = B * L * di;
    if (idx >= total) return;
    int ch = idx % di;
    int l  = (idx / di) % L;
    int b  = idx / (di * L);
    const float* base = xz + (size_t)b * L * ldxz + ch;
    float acc = bias[ch];
#pragma unroll
    for (int k = 0; k < 4; ++k) {
        int ls = rev ? (l + 3 - k) : (l - 3 + k);
        if (ls >= 0 && ls < L) acc += base[(size_t)ls * ldxz] * w[ch * 4 + k];
    }
    xc[idx] = siluf_(acc);
}

// selective scan: 16 lanes per (batch, channel); lane n carries state n,
// y-reduction via 4 shfl_xor steps at width 16 (wave32 holds 2 groups).
__global__ void scan_kernel(float* __restrict__ y, const float* __restrict__ delta,
                            const float* __restrict__ xdb, int ldx, int dtr,
                            const float* __restrict__ xc, const float* __restrict__ A_log,
                            int B, int L, int di, int rev)
{
    int gid = blockIdx.x * blockDim.x + threadIdx.x;
    int n   = gid & 15;
    int grp = gid >> 4;
    if (grp >= B * di) return;
    int d = grp % di, b = grp / di;
    float Aa = -__expf(A_log[d * 16 + n]);
    float h  = 0.f;
    for (int t = 0; t < L; ++t) {
        int l = rev ? (L - 1 - t) : t;
        size_t m = (size_t)b * L + l;
        float dl = delta[m * di + d];
        float u  = xc[m * di + d];
        float bn = xdb[m * ldx + dtr + n];
        float cn = xdb[m * ldx + dtr + 16 + n];
        h = __expf(dl * Aa) * h + dl * bn * u;
        float yv = h * cn;
#pragma unroll
        for (int o = 8; o > 0; o >>= 1) yv += __shfl_xor(yv, o, 16);
        if (n == 0) y[m * di + d] = yv;
    }
}

// y = (y + xc*D) * silu(z)
__global__ void gate_kernel(float* __restrict__ y, const float* __restrict__ xc,
                            const float* __restrict__ Dp, const float* __restrict__ xz,
                            int ldxz, int di_off, int M, int di)
{
    int idx = blockIdx.x * blockDim.x + threadIdx.x;
    if (idx >= M * di) return;
    int ch = idx % di;
    int m  = idx / di;
    float z = xz[(size_t)m * ldxz + di_off + ch];
    y[idx] = (y[idx] + xc[idx] * Dp[ch]) * siluf_(z);
}

// (B,H,W,C) -> (B, H/2*W/2, 4C), concat order [ee, oe, eo, oo]
__global__ void merge_gather_kernel(float* __restrict__ out, const float* __restrict__ in,
                                    int B, int H, int Wd, int C)
{
    int idx = blockIdx.x * blockDim.x + threadIdx.x;
    int Wh = Wd / 2, Hh = H / 2;
    int total = B * Hh * Wh * 4 * C;
    if (idx >= total) return;
    int c4 = idx % (4 * C);
    int p  = (idx / (4 * C)) % (Hh * Wh);
    int b  = idx / (4 * C * Hh * Wh);
    int q = c4 / C, c = c4 % C;
    int dy = q & 1, dx = q >> 1;
    int ph = p / Wh, pw = p % Wh;
    int hy = 2 * ph + dy, wx = 2 * pw + dx;
    out[idx] = in[(((size_t)b * H + hy) * Wd + wx) * C + c];
}

__global__ void mean_pool_kernel(float* __restrict__ feat, const float* __restrict__ x,
                                 int B, int L, int C)
{
    int idx = blockIdx.x * blockDim.x + threadIdx.x;
    if (idx >= B * C) return;
    int c = idx % C, b = idx / C;
    float s = 0.f;
    for (int l = 0; l < L; ++l) s += x[((size_t)b * L + l) * C + c];
    feat[idx] = s / L;
}

// ---------------------------------------------------------------- orchestration
extern "C" void kernel_launch(void* const* d_in, const int* in_sizes, int n_in,
                              void* d_out, int out_size, void* d_ws, size_t ws_size,
                              hipStream_t stream)
{
    (void)in_sizes; (void)n_in; (void)out_size; (void)ws_size;

    int idx = 0;
    auto F = [&](void) { return (const float*)d_in[idx++]; };

    const float* x      = F();
    const float* pe_w   = F();
    const float* pe_b   = F();
    const float* pe_lnw = F();
    const float* pe_lnb = F();

    struct BlockP { const float *norm_w, *in_proj, *conv_w, *conv_b, *x_proj,
                               *dt_w, *dt_b, *A_log, *D, *out_proj; };
    struct MergeP { const float *ln_w, *ln_b, *red_w; };
    const int depths[4] = {2, 2, 9, 2};
    BlockP blocks[4][9];
    MergeP merges[3];
    for (int s = 0; s < 4; ++s) {
        for (int bi = 0; bi < depths[s]; ++bi) {
            BlockP& bp = blocks[s][bi];
            bp.norm_w = F(); bp.in_proj = F(); bp.conv_w = F(); bp.conv_b = F();
            bp.x_proj = F(); bp.dt_w = F();    bp.dt_b = F();   bp.A_log = F();
            bp.D = F();      bp.out_proj = F();
        }
        if (s < 3) { merges[s].ln_w = F(); merges[s].ln_b = F(); merges[s].red_w = F(); }
    }
    const float* fin_lnw = F();
    const float* fin_lnb = F();
    const float* head_w  = F();
    const float* head_b  = F();

    const int Bsz = 8;

    // bump-allocated workspace (floats); sized for stage 0 maxima
    float* ws = (float*)d_ws;
    size_t off = 0;
    auto alloc = [&](size_t n) { float* p = ws + off; off += n; return p; };
    float* cur   = alloc(2600000);   // activations (B*L*dim)
    float* res   = alloc(2600000);   // residual stream
    float* hnorm = alloc(2600000);   // rmsnorm output
    float* xz    = alloc(9700000);   // in_proj output (B*L, 2*di)
    float* xcb   = alloc(4900000);   // conv+silu output (B*L, di)
    float* xdb   = alloc(1100000);   // x_proj output (B*L, ldxp padded)
    float* dlt   = alloc(4900000);   // softplus delta (B*L, di)
    float* ybuf  = alloc(4900000);   // scan output (B*L, di)
    float* mrg   = alloc(2600000);   // patch-merge gather (B*L/4, 4C)
    float* feat  = alloc(8192);      // pooled features (B, 768)

    // ---- patch embed + LN
    {
        int total = Bsz * 3136 * 96;
        patch_embed_kernel<<<(total + 255) / 256, 256, 0, stream>>>(cur, x, pe_w, pe_b, Bsz);
        layernorm_kernel<<<Bsz * 3136, 32, 0, stream>>>(cur, pe_lnw, pe_lnb, Bsz * 3136, 96);
    }

    int Hh = 56, Ww = 56, dim = 96;
    for (int s = 0; s < 4; ++s) {
        int L    = Hh * Ww;
        int M    = Bsz * L;
        int di   = 2 * dim;
        int dtr  = (dim + 15) / 16;
        int nxdb = dtr + 32;               // valid columns of x_proj output
        int ldxp = (nxdb + 3) & ~3;        // padded row stride (16B-aligned rows)
        for (int bi = 0; bi < depths[s]; ++bi) {
            const BlockP& p = blocks[s][bi];
            int n = M * dim;
            residual_kernel<<<(n + 255) / 256, 256, 0, stream>>>(res, cur, n, bi == 0 ? 1 : 0);
            rmsnorm_kernel<<<M, 32, 0, stream>>>(hnorm, res, p.norm_w, M, dim);
            // shared per-token projection (flip-invariant)
            gemm(stream, xz, 2 * di, hnorm, dim, p.in_proj, nullptr, M, 2 * di, dim, 0, false);
            for (int rev = 0; rev < 2; ++rev) {
                int nt = M * di;
                conv_silu_kernel<<<(nt + 255) / 256, 256, 0, stream>>>(
                    xcb, xz, 2 * di, p.conv_w, p.conv_b, Bsz, L, di, rev);
                gemm(stream, xdb, ldxp, xcb, di, p.x_proj, nullptr, M, nxdb, di, 0, false);
                // delta = softplus(dt @ dt_w^T + dt_b); dt = first dtr cols of xdb
                gemm(stream, dlt, di, xdb, ldxp, p.dt_w, p.dt_b, M, di, dtr, 1, false);
                scan_kernel<<<(Bsz * di * 16 + 255) / 256, 256, 0, stream>>>(
                    ybuf, dlt, xdb, ldxp, dtr, xcb, p.A_log, Bsz, L, di, rev);
                gate_kernel<<<(nt + 255) / 256, 256, 0, stream>>>(
                    ybuf, xcb, p.D, xz, 2 * di, di, M, di);
                // fwd writes, bwd accumulates: cur = y_f@out_proj^T + y_b@out_proj^T
                gemm(stream, cur, dim, ybuf, di, p.out_proj, nullptr, M, dim, di, 0, rev == 1);
            }
        }
        if (s < 3) {
            int M2 = Bsz * (Hh / 2) * (Ww / 2);
            int C4 = 4 * dim;
            int nt = M2 * C4;
            merge_gather_kernel<<<(nt + 255) / 256, 256, 0, stream>>>(mrg, cur, Bsz, Hh, Ww, dim);
            layernorm_kernel<<<M2, 32, 0, stream>>>(mrg, merges[s].ln_w, merges[s].ln_b, M2, C4);
            gemm(stream, cur, 2 * dim, mrg, C4, merges[s].red_w, nullptr, M2, 2 * dim, C4, 0, false);
            Hh /= 2; Ww /= 2; dim *= 2;
        }
    }

    // ---- final LN, mean pool, head
    {
        int L = Hh * Ww;
        int M = Bsz * L;
        layernorm_kernel<<<M, 32, 0, stream>>>(cur, fin_lnw, fin_lnb, M, dim);
        mean_pool_kernel<<<(Bsz * dim + 255) / 256, 256, 0, stream>>>(feat, cur, Bsz, L, dim);
        gemm(stream, (float*)d_out, 1000, feat, dim, head_w, head_b, Bsz, 1000, dim, 0, false);
    }
}
